// GRUMulitiTaskAutoencoder_38929583571515
// MI455X (gfx1250) — matrix operand order repaired and checked by reading, not compiled, not run on hardware
//
#include <hip/hip_runtime.h>

typedef __attribute__((ext_vector_type(16))) _Float16 v16h;
typedef __attribute__((ext_vector_type(8)))  float    v8f;
typedef __attribute__((ext_vector_type(4)))  unsigned int v4u;
typedef __attribute__((ext_vector_type(4)))  int v4i;
typedef __attribute__((ext_vector_type(8)))  int v8i;

#define H 128
#define SEQ 128
#define NE 8
#define NROWS 2048
#define MTILE 16
#define HPAD 136                    // padded halfs per LDS h row (bank-conflict free)
#define ENC_HALFS (3*H*H)           // 49152
#define DEC_HALFS (NE*3*H*H)        // 393216
#define LIN_HALFS (NE*H)            // 1024

__device__ __forceinline__ float sigm(float x)      { return 1.0f / (1.0f + __expf(-x)); }
__device__ __forceinline__ float tanh_fast(float x) { return 2.0f / (1.0f + __expf(-2.0f * x)) - 1.0f; }

union Frag16 { uint4 q[2]; v16h h; };

// A fragment (16x32 f16): lane m = row, 16 contiguous K halfs at c*32+16*hf
__device__ __forceinline__ v16h load_a(const _Float16* hcur, int m, int c, int hf) {
  Frag16 f;
  const _Float16* p = hcur + m * HPAD + c * 32 + 16 * hf;
  f.q[0] = *(const uint4*)p;
  f.q[1] = *(const uint4*)(p + 8);
  return f.h;
}
// B fragment (32x16 f16): lane = column g, 16 contiguous K halfs at 16*hf
__device__ __forceinline__ v16h load_b(const _Float16* w) {
  Frag16 f;
  f.q[0] = *(const uint4*)w;
  f.q[1] = *(const uint4*)(w + 8);
  return f.h;
}

__device__ __forceinline__ v8f wmma16(v16h a, v16h b, v8f c) {
  return __builtin_amdgcn_wmma_f32_16x16x32_f16(false, a, false, b, (short)0, c, false, false);
}

__device__ __forceinline__ v8f splat8(float v) {
  v8f r;
#pragma unroll
  for (int i = 0; i < 8; ++i) r[i] = v;
  return r;
}

__global__ void prep_weights(const float* __restrict__ enc, const float* __restrict__ dec,
                             const float* __restrict__ lin, _Float16* __restrict__ ws, int total) {
  int i = blockIdx.x * blockDim.x + threadIdx.x;
  if (i >= total) return;
  float v;
  if (i < ENC_HALFS)                 v = enc[i];
  else if (i < ENC_HALFS + DEC_HALFS) v = dec[i - ENC_HALFS];
  else                                v = lin[i - ENC_HALFS - DEC_HALFS];
  ws[i] = (_Float16)v;
}

__global__ __launch_bounds__(256)
void gru_autoenc_kernel(const float* __restrict__ x,
                        const float* __restrict__ WihE, const float* __restrict__ bihE,
                        const float* __restrict__ bhhE,
                        const float* __restrict__ WihD, const float* __restrict__ bihD,
                        const float* __restrict__ bhhD,
                        const float* __restrict__ blin,
                        const _Float16* __restrict__ wsE, const _Float16* __restrict__ wsD,
                        const _Float16* __restrict__ wsL,
                        float* __restrict__ out) {
  __shared__ float    x_tile[MTILE * 129];
  __shared__ _Float16 hbuf[2][MTILE * HPAD];

  const int tid  = threadIdx.x;
  const int lane = tid & 31;
  const int w    = tid >> 5;       // wave 0..7 -> owns output cols [16w,16w+16)
  const int n16  = lane & 15;      // column within 16-wide tile / D-matrix col
  const int hf   = lane >> 4;      // half-wave selector
  const int col  = w * 16 + n16;   // 0..127
  const int row0 = blockIdx.x * MTILE;

  // ---- Stage x tile (16 rows x 129 f32) into LDS via Tensor Data Mover ----
  if (w == 0) {
    unsigned long long gaddr = (unsigned long long)(const void*)(x + (unsigned long long)row0 * 129ull);
    unsigned int laddr = (unsigned int)(unsigned long long)(const void*)&x_tile[0];
    v4u g0; v8i g1; v4i z4 = {0, 0, 0, 0}; v8i z8 = {0, 0, 0, 0, 0, 0, 0, 0};
    g0[0] = 1u;                                                    // count=1, user mode
    g0[1] = laddr;                                                 // lds_addr
    g0[2] = (unsigned int)(gaddr & 0xFFFFFFFFull);                 // global_addr lo
    g0[3] = (unsigned int)((gaddr >> 32) & 0x1FFFFFFull) | (2u << 30); // addr hi | type=2
    g1[0] = (int)(2u << 16);                                       // data_size = 4B
    g1[1] = (int)(129u << 16);                                     // tensor_dim0 lo16
    g1[2] = (int)(2048u << 16);                                    // dim0 hi16=0 | tensor_dim1 lo16
    g1[3] = (int)(129u << 16);                                     // dim1 hi16=0 | tile_dim0=129
    g1[4] = (int)16u;                                              // tile_dim1=16, tile_dim2=0
    g1[5] = (int)129u;                                             // tensor_dim0_stride lo32
    g1[6] = 0; g1[7] = 0;
#if __clang_major__ >= 23
    __builtin_amdgcn_tensor_load_to_lds(g0, g1, z4, z4, z8, 0);
#else
    __builtin_amdgcn_tensor_load_to_lds(g0, g1, z4, z4, 0);
#endif
    __builtin_amdgcn_s_wait_tensorcnt(0);
  }
  for (int i = tid; i < MTILE * HPAD; i += 256) hbuf[0][i] = (_Float16)0.0f;
  __syncthreads();

  // ---- Encoder per-lane constants ----
  const float wihR = WihE[col], wihZ = WihE[128 + col], wihN = WihE[256 + col];
  const float bihR = bihE[col], bihZ = bihE[128 + col], bihN = bihE[256 + col];
  const float bhhR = bhhE[col], bhhZ = bhhE[128 + col], bhhN = bhhE[256 + col];

  // Encoder W_hh B fragments: register-resident for all 128 steps (96 VGPRs)
  v16h Ber[4], Bez[4], Ben[4];
#pragma unroll
  for (int c = 0; c < 4; ++c) {
    Ber[c] = load_b(wsE + (col)*H       + c * 32 + 16 * hf);
    Bez[c] = load_b(wsE + (128 + col)*H + c * 32 + 16 * hf);
    Ben[c] = load_b(wsE + (256 + col)*H + c * 32 + 16 * hf);
  }

  float hc[8];
#pragma unroll
  for (int r = 0; r < 8; ++r) hc[r] = 0.0f;

  // ---- Encoder: 128 steps ----
  for (int t = 0; t < SEQ; ++t) {
    const _Float16* hcur = hbuf[t & 1];
    _Float16* hnxt = hbuf[(t + 1) & 1];
    v16h a[4];
#pragma unroll
    for (int c = 0; c < 4; ++c) a[c] = load_a(hcur, n16, c, hf);
    v8f aR = splat8(bhhR), aZ = splat8(bhhZ), aN = splat8(bhhN);
#pragma unroll
    for (int c = 0; c < 4; ++c) {
      aR = wmma16(a[c], Ber[c], aR);
      aZ = wmma16(a[c], Bez[c], aZ);
      aN = wmma16(a[c], Ben[c], aN);
    }
    float xl = x_tile[n16 * 129 + t];
#pragma unroll
    for (int r = 0; r < 8; ++r) {
      int rowr = r + 8 * hf;
      float xr = __shfl(xl, rowr, 32);
      float rg = sigm(fmaf(xr, wihR, bihR) + aR[r]);
      float zg = sigm(fmaf(xr, wihZ, bihZ) + aZ[r]);
      float ng = tanh_fast(fmaf(xr, wihN, bihN) + rg * aN[r]);
      float hn = (1.0f - zg) * ng + zg * hc[r];
      hc[r] = hn;
      hnxt[rowr * HPAD + col] = (_Float16)hn;
    }
    __syncthreads();
  }

  // ---- Decoder setup: expert range of this (sorted-id) tile ----
  int idl = (int)x_tile[n16 * 129 + 128];
  int emin = idl, emax = idl;
#pragma unroll
  for (int m = 1; m < 16; m <<= 1) {
    emin = min(emin, __shfl_xor(emin, m, 32));
    emax = max(emax, __shfl_xor(emax, m, 32));
  }
  const int e0 = __builtin_amdgcn_readfirstlane(emin);
  const int e1 = __builtin_amdgcn_readfirstlane(emax);

  int   idv[8];
  float WiR[8], WiZ[8], WiN[8], biR[8], biZ[8], biN[8];
  v8f bgR, bgZ, bgN;
#pragma unroll
  for (int r = 0; r < 8; ++r) {
    int rowr = r + 8 * hf;
    int e = (int)x_tile[rowr * 129 + 128];
    idv[r] = e;
    WiR[r] = WihD[e * 384 + col];       WiZ[r] = WihD[e * 384 + 128 + col]; WiN[r] = WihD[e * 384 + 256 + col];
    biR[r] = bihD[e * 384 + col];       biZ[r] = bihD[e * 384 + 128 + col]; biN[r] = bihD[e * 384 + 256 + col];
    bgR[r] = bhhD[e * 384 + col];       bgZ[r] = bhhD[e * 384 + 128 + col]; bgN[r] = bhhD[e * 384 + 256 + col];
  }
  // y-WMMA B fragments: column n = tile row n, weights W_lin[id[n],:] (32 VGPRs)
  v16h Bly[4];
#pragma unroll
  for (int c = 0; c < 4; ++c) Bly[c] = load_b(wsL + idl * H + c * 32 + 16 * hf);

  // diagonal-holder bookkeeping: D[m][m] lives in lane m (m<8, r=m) and lane m+16 (m>=8, r=m-8)
  const bool holder  = (lane < 8) || (lane >= 24);
  const int  rowHeld = (lane < 8) ? lane : ((lane >= 24) ? (lane - 16) : 0);
  const int  hsel    = ((lane < 8) ? lane : (lane - 24)) & 7;
  const float blHold = blin[(int)x_tile[rowHeld * 129 + 128]];

  float yprev[8];
#pragma unroll
  for (int r = 0; r < 8; ++r) yprev[r] = 0.0f;

  // ---- Decoder: 128 steps; y^(t) computed from h^(t) via WMMA diagonal ----
  for (int t = 0; t < SEQ; ++t) {
    const _Float16* hcur = hbuf[t & 1];
    _Float16* hnxt = hbuf[(t + 1) & 1];
    v16h a[4];
#pragma unroll
    for (int c = 0; c < 4; ++c) a[c] = load_a(hcur, n16, c, hf);

    if (t > 0) {  // y^(t) = diag(h^(t) x Wl^T) + bl ; y^(0) = 0 (peeled)
      v8f yacc = splat8(0.0f);
#pragma unroll
      for (int c = 0; c < 4; ++c) yacc = wmma16(a[c], Bly[c], yacc);
      float yv = yacc[hsel] + blHold;
      if (w == 0 && holder) out[(row0 + rowHeld) * SEQ + (t - 1)] = yv;
#pragma unroll
      for (int r = 0; r < 8; ++r) yprev[r] = __shfl(yv, r + 24 * hf, 32);
    }

    v8f aR, aZ, aN;
    for (int e = e0; e <= e1; ++e) {
      v8f tR = bgR, tZ = bgZ, tN = bgN;
      const _Float16* wd = wsD + e * (3 * H * H);
#pragma unroll
      for (int c = 0; c < 4; ++c) {
        tR = wmma16(a[c], load_b(wd + (col)*H       + c * 32 + 16 * hf), tR);
        tZ = wmma16(a[c], load_b(wd + (128 + col)*H + c * 32 + 16 * hf), tZ);
        tN = wmma16(a[c], load_b(wd + (256 + col)*H + c * 32 + 16 * hf), tN);
      }
      if (e == e0) { aR = tR; aZ = tZ; aN = tN; }
      else {
#pragma unroll
        for (int r = 0; r < 8; ++r)
          if (idv[r] == e) { aR[r] = tR[r]; aZ[r] = tZ[r]; aN[r] = tN[r]; }
      }
    }
#pragma unroll
    for (int r = 0; r < 8; ++r) {
      float rg = sigm(fmaf(yprev[r], WiR[r], biR[r]) + aR[r]);
      float zg = sigm(fmaf(yprev[r], WiZ[r], biZ[r]) + aZ[r]);
      float ng = tanh_fast(fmaf(yprev[r], WiN[r], biN[r]) + rg * aN[r]);
      float hn = (1.0f - zg) * ng + zg * hc[r];
      hc[r] = hn;
      hnxt[(r + 8 * hf) * HPAD + col] = (_Float16)hn;
    }
    __syncthreads();
  }

  // ---- Epilogue: y^(SEQ) from h^(SEQ) -> output column SEQ-1 ----
  {
    const _Float16* hcur = hbuf[SEQ & 1];
    v8f yacc = splat8(0.0f);
#pragma unroll
    for (int c = 0; c < 4; ++c) yacc = wmma16(load_a(hcur, n16, c, hf), Bly[c], yacc);
    float yv = yacc[hsel] + blHold;
    if (w == 0 && holder) out[(row0 + rowHeld) * SEQ + (SEQ - 1)] = yv;
  }
}

extern "C" void kernel_launch(void* const* d_in, const int* in_sizes, int n_in,
                              void* d_out, int out_size, void* d_ws, size_t ws_size,
                              hipStream_t stream) {
  const float* x    = (const float*)d_in[0];
  const float* WihE = (const float*)d_in[1];
  const float* WhhE = (const float*)d_in[2];
  const float* bihE = (const float*)d_in[3];
  const float* bhhE = (const float*)d_in[4];
  const float* WihD = (const float*)d_in[5];
  const float* WhhD = (const float*)d_in[6];
  const float* bihD = (const float*)d_in[7];
  const float* bhhD = (const float*)d_in[8];
  const float* Wlin = (const float*)d_in[9];
  const float* blin = (const float*)d_in[10];
  float* out = (float*)d_out;
  _Float16* wsE = (_Float16*)d_ws;
  _Float16* wsD = wsE + ENC_HALFS;
  _Float16* wsL = wsD + DEC_HALFS;

  int total = ENC_HALFS + DEC_HALFS + LIN_HALFS;
  prep_weights<<<(total + 255) / 256, 256, 0, stream>>>(WhhE, WhhD, Wlin, wsE, total);
  gru_autoenc_kernel<<<NROWS / MTILE, 256, 0, stream>>>(
      x, WihE, bihE, bhhE, WihD, bihD, bhhD, blin, wsE, wsD, wsL, out);
}
